// TriangleSelfAttention_30580167147686
// MI455X (gfx1250) — matrix operand order repaired and checked by reading, not compiled, not run on hardware
//
#include <hip/hip_runtime.h>
#include <hip/hip_bf16.h>

typedef _Float16 f16;
typedef __attribute__((ext_vector_type(16))) _Float16 v16h;
typedef __attribute__((ext_vector_type(8)))  float    v8f;
typedef __attribute__((ext_vector_type(4)))  int      v4i;

#define IN_C  256
#define OUT_C 256
#define NH    8
#define DH    32
#define LW    192
#define NTOK  (LW * LW)   // 36864

// Padded LDS row strides (halves) so 16 lanes hit 16 distinct banks on b128 reads
#define WSTRIDE  (256 + 8)   // GEMM weight tile rows
#define KSTRIDE  (DH + 8)    // attention K rows (32 -> 40)
#define PSTRIDE  (LW + 8)    // attention P / Vt rows (192 -> 200)

// ---------------------------------------------------------------------------
// Async global->LDS staging (CDNA5 ASYNCcnt path) with compile-safe fallback
// Builtin signature (per hipcc diagnostic): (v4i AS1*, v4i AS3*, imm, imm)
// ---------------------------------------------------------------------------
#if defined(__has_builtin)
#if __has_builtin(__builtin_amdgcn_global_load_async_to_lds_b128) && \
    __has_builtin(__builtin_amdgcn_s_wait_asynccnt)
#define HAVE_ASYNC_LDS 1
#endif
#endif

#ifdef HAVE_ASYNC_LDS
typedef v4i __attribute__((address_space(1))) gv4i;
typedef v4i __attribute__((address_space(3))) lv4i;
#endif

__device__ inline void lds_copy16(f16* dst_lds, const f16* src_global) {
#ifdef HAVE_ASYNC_LDS
  gv4i* g = reinterpret_cast<gv4i*>(reinterpret_cast<uintptr_t>(src_global));
  lv4i* l = reinterpret_cast<lv4i*>(
      static_cast<uint32_t>(reinterpret_cast<uintptr_t>(dst_lds)));
  __builtin_amdgcn_global_load_async_to_lds_b128(g, l, 0, 0);
#else
  *(uint4*)dst_lds = *(const uint4*)src_global;
#endif
}

__device__ inline void lds_stage_fence() {
#ifdef HAVE_ASYNC_LDS
  __builtin_amdgcn_s_wait_asynccnt(0);
#endif
  __syncthreads();
}

// ---------------------------------------------------------------------------
// Fragment loaders (CDNA5 V_WMMA_F32_16X16X32_F16 layouts, wave32)
// A 16x32 f16: lane m = lane&15; kbase = (lane<16)?0:8;
//   elems 0..7 -> K = kbase..kbase+7 ; elems 8..15 -> K = kbase+16..kbase+23
// B 32x16 f16 from row-major [N][K]: lane n = lane&15; kbase = (lane<16)?0:8;
//   elems 0..7 -> K = kbase..kbase+7 ; elems 8..15 -> K = kbase+16..kbase+23
// C/D f32: vgpr r -> row = r + (lane<16?0:8); col = lane&15
// ---------------------------------------------------------------------------
__device__ inline v16h load_a_frag(const f16* __restrict__ A, int row_stride,
                                   int m_base, int k0, int lane) {
  int m  = m_base + (lane & 15);
  int kb = k0 + ((lane < 16) ? 0 : 8);
  const f16* p = A + (size_t)m * row_stride + kb;
  v16h a;
  *((uint4*)&a)       = *((const uint4*)p);
  *(((uint4*)&a) + 1) = *((const uint4*)(p + 16));
  return a;
}

__device__ inline v16h load_b_frag(const f16* __restrict__ Wr, int row_stride,
                                   int n_base, int k0, int lane) {
  int n  = n_base + (lane & 15);
  int kb = k0 + ((lane < 16) ? 0 : 8);
  const f16* p = Wr + (size_t)n * row_stride + kb;
  v16h b;
  *((uint4*)&b)       = *((const uint4*)p);
  *(((uint4*)&b) + 1) = *((const uint4*)(p + 16));
  return b;
}

// ---------------------------------------------------------------------------
// LayerNorm: one wave per token (wave32: 8 channels/lane), f16 output
// ---------------------------------------------------------------------------
__global__ __launch_bounds__(256) void ln_kernel(
    const float* __restrict__ x, const float* __restrict__ w,
    const float* __restrict__ b, f16* __restrict__ xn, int ntok) {
  int lane = threadIdx.x & 31;
  int wave = threadIdx.x >> 5;
  int tok  = blockIdx.x * 8 + wave;
  if (tok >= ntok) return;
  const float* p = x + (size_t)tok * IN_C + lane * 8;
  float4 v0 = *(const float4*)p;
  float4 v1 = *(const float4*)(p + 4);
  float s  = v0.x + v0.y + v0.z + v0.w + v1.x + v1.y + v1.z + v1.w;
  float ss = v0.x*v0.x + v0.y*v0.y + v0.z*v0.z + v0.w*v0.w +
             v1.x*v1.x + v1.y*v1.y + v1.z*v1.z + v1.w*v1.w;
  for (int m = 1; m <= 16; m <<= 1) {
    s  += __shfl_xor(s,  m, 32);
    ss += __shfl_xor(ss, m, 32);
  }
  float mu  = s * (1.0f / IN_C);
  float var = ss * (1.0f / IN_C) - mu * mu;
  float rs  = rsqrtf(var + 1e-5f);
  float4 w0 = *(const float4*)(w + lane * 8);
  float4 w1 = *(const float4*)(w + lane * 8 + 4);
  float4 b0 = *(const float4*)(b + lane * 8);
  float4 b1 = *(const float4*)(b + lane * 8 + 4);
  f16 o[8];
  o[0] = (f16)((v0.x - mu) * rs * w0.x + b0.x);
  o[1] = (f16)((v0.y - mu) * rs * w0.y + b0.y);
  o[2] = (f16)((v0.z - mu) * rs * w0.z + b0.z);
  o[3] = (f16)((v0.w - mu) * rs * w0.w + b0.w);
  o[4] = (f16)((v1.x - mu) * rs * w1.x + b1.x);
  o[5] = (f16)((v1.y - mu) * rs * w1.y + b1.y);
  o[6] = (f16)((v1.z - mu) * rs * w1.z + b1.z);
  o[7] = (f16)((v1.w - mu) * rs * w1.w + b1.w);
  *(uint4*)(xn + (size_t)tok * IN_C + lane * 8) = *(uint4*)o;
}

// ---------------------------------------------------------------------------
// Elementwise helpers
// ---------------------------------------------------------------------------
__global__ __launch_bounds__(256) void cvt_f16_kernel(
    const float* __restrict__ src, f16* __restrict__ dst, int n) {
  int i = blockIdx.x * blockDim.x + threadIdx.x;
  if (i < n) dst[i] = (f16)src[i];
}

// coef = (1/sqrt(DH)) * exp(-(d/8)^2 / 2)   (softmax scale folded in)
__global__ __launch_bounds__(256) void coef_kernel(
    const float* __restrict__ d, float* __restrict__ c, int n) {
  int i = blockIdx.x * blockDim.x + threadIdx.x;
  if (i < n) {
    float x = d[i];
    c[i] = 0.1767766952966369f * __expf(-x * x * (1.0f / 128.0f));
  }
}

// ---------------------------------------------------------------------------
// WMMA GEMM: C[M,N] = A_f16[M,K] * Wr_f16[N,K]^T + bias
// 8 waves / block; wave computes a 16x64 tile. Weight tile staged in LDS
// (async when available), A fragments software-pipelined. K <= 256.
// ---------------------------------------------------------------------------
__global__ __launch_bounds__(256) void gemm_wmma_kernel(
    const f16* __restrict__ A, const f16* __restrict__ Wr,
    const float* __restrict__ bias,
    f16* __restrict__ out16, float* __restrict__ out32,
    int M, int N, int K) {
  __shared__ f16 ldsW[64 * WSTRIDE];   // 64 weight rows, padded stride

  const int tid  = threadIdx.x;
  const int lane = tid & 31;
  const int wave = tid >> 5;
  const int m0 = blockIdx.x * 128 + wave * 16;
  const int n0 = blockIdx.y * 64;

  // Stage 64 x K weight tile into LDS, 16B chunks
  {
    const f16* wsrc = Wr + (size_t)n0 * K;
    const int chunks = 64 * (K / 8);
    for (int u = tid; u < chunks; u += 256) {
      int r  = u / (K / 8);
      int ck = (u % (K / 8)) * 8;
      lds_copy16(&ldsW[r * WSTRIDE + ck], wsrc + (size_t)r * K + ck);
    }
  }
  lds_stage_fence();

  v8f acc[4] = {};
  v16h a = load_a_frag(A, K, m0, 0, lane);
  for (int k0 = 0; k0 < K; k0 += 32) {
    int kn = (k0 + 32 < K) ? (k0 + 32) : k0;
    v16h a_next = load_a_frag(A, K, m0, kn, lane);   // prefetch next K-step
#pragma unroll
    for (int nt = 0; nt < 4; ++nt) {
      v16h b = load_b_frag(ldsW, WSTRIDE, nt * 16, k0, lane);
      acc[nt] = __builtin_amdgcn_wmma_f32_16x16x32_f16(
          false, a, false, b, (short)0, acc[nt], false, false);
    }
    a = a_next;
  }

  const int col   = lane & 15;
  const int rbase = (lane < 16) ? 0 : 8;
  if (out16) {
#pragma unroll
    for (int nt = 0; nt < 4; ++nt) {
      int n = n0 + nt * 16 + col;
      float bv = bias ? bias[n] : 0.0f;
#pragma unroll
      for (int r = 0; r < 8; ++r)
        out16[(size_t)(m0 + rbase + r) * N + n] = (f16)(acc[nt][r] + bv);
    }
  } else {
#pragma unroll
    for (int nt = 0; nt < 4; ++nt) {
      int n = n0 + nt * 16 + col;
      float bv = bias ? bias[n] : 0.0f;
#pragma unroll
      for (int r = 0; r < 8; ++r)
        out32[(size_t)(m0 + rbase + r) * N + n] = acc[nt][r] + bv;
    }
  }
}

// ---------------------------------------------------------------------------
// Attention: one block per (row n, head h). 4 waves, each owns 16 query rows
// per outer iteration (192 rows -> 3 iterations/wave).
// S = (q k^T) * coefs ; exact softmax ; O = P V ; out = O * gate (f16)
// V is staged TRANSPOSED (vT[c][j]) so P x V B-fragments are contiguous.
// ---------------------------------------------------------------------------
#define ATT_WAVES 4
__global__ __launch_bounds__(ATT_WAVES * 32) void attn_kernel(
    const f16* __restrict__ qf, const f16* __restrict__ kf,
    const f16* __restrict__ vf, const f16* __restrict__ gf,
    const float* __restrict__ coefs, f16* __restrict__ out) {
  __shared__ f16 ldsK[LW * KSTRIDE];               // 15.0 KB
  __shared__ f16 ldsVT[DH * PSTRIDE];              // 12.5 KB
  __shared__ f16 ldsP[ATT_WAVES * 16 * PSTRIDE];   // 25.0 KB

  const int n = blockIdx.x;   // Lr row
  const int h = blockIdx.y;   // head
  const int tid = threadIdx.x;
  const int lane = tid & 31, wave = tid >> 5;
  const int col = lane & 15;
  const int rbase = (lane < 16) ? 0 : 8;

  // Stage K (async 16B chunks) into padded rows
  for (int u = tid; u < LW * 2; u += blockDim.x) {
    int j = u >> 1, cc = (u & 1) * 16;
    lds_copy16(&ldsK[j * KSTRIDE + cc],
               kf + ((size_t)(n * LW + j)) * OUT_C + h * DH + cc);
  }
  // Stage V transposed: vT[c][j] = V[j][c]
  for (int u = tid; u < LW * (DH / 2); u += blockDim.x) {
    int j = u / (DH / 2), c = (u % (DH / 2)) * 2;
    const f16* src = vf + ((size_t)(n * LW + j)) * OUT_C + h * DH + c;
    f16 x0 = src[0], x1 = src[1];
    ldsVT[c * PSTRIDE + j]       = x0;
    ldsVT[(c + 1) * PSTRIDE + j] = x1;
  }
  lds_stage_fence();

  for (int it = wave; it < LW / 16; it += ATT_WAVES) {
    int i0 = it * 16;

    // A fragment of q (K = 32, single chunk)
    v16h aq;
    {
      int m  = i0 + (lane & 15);
      int kb = (lane < 16) ? 0 : 8;
      const f16* p = qf + ((size_t)(n * LW + m)) * OUT_C + h * DH + kb;
      *((uint4*)&aq)       = *((const uint4*)p);
      *(((uint4*)&aq) + 1) = *((const uint4*)(p + 16));
    }

    // Scores: 12 tiles of 16x16
    v8f s[12];
#pragma unroll
    for (int jt = 0; jt < 12; ++jt) {
      v16h b = load_b_frag(ldsK, KSTRIDE, jt * 16, 0, lane);
      v8f z = {};
      s[jt] = __builtin_amdgcn_wmma_f32_16x16x32_f16(
          false, aq, false, b, (short)0, z, false, false);
    }

    // coef (scale folded in), row max
    float rmax[8];
#pragma unroll
    for (int r = 0; r < 8; ++r) rmax[r] = -3.0e38f;
#pragma unroll
    for (int jt = 0; jt < 12; ++jt) {
#pragma unroll
      for (int r = 0; r < 8; ++r) {
        int i = i0 + rbase + r;
        int j = jt * 16 + col;
        float v = s[jt][r] * coefs[i * LW + j];
        s[jt][r] = v;
        rmax[r] = fmaxf(rmax[r], v);
      }
    }
#pragma unroll
    for (int r = 0; r < 8; ++r) {
      float v = rmax[r];                     // reduce within 16-lane half
      v = fmaxf(v, __shfl_xor(v, 1, 32));
      v = fmaxf(v, __shfl_xor(v, 2, 32));
      v = fmaxf(v, __shfl_xor(v, 4, 32));
      v = fmaxf(v, __shfl_xor(v, 8, 32));
      rmax[r] = v;
    }

    float rsum[8];
#pragma unroll
    for (int r = 0; r < 8; ++r) rsum[r] = 0.0f;
#pragma unroll
    for (int jt = 0; jt < 12; ++jt) {
#pragma unroll
      for (int r = 0; r < 8; ++r) {
        float e = __expf(s[jt][r] - rmax[r]);
        s[jt][r] = e;
        rsum[r] += e;
      }
    }
#pragma unroll
    for (int r = 0; r < 8; ++r) {
      float v = rsum[r];
      v += __shfl_xor(v, 1, 32);
      v += __shfl_xor(v, 2, 32);
      v += __shfl_xor(v, 4, 32);
      v += __shfl_xor(v, 8, 32);
      rsum[r] = 1.0f / v;
    }

    // Write normalized P (f16) to this wave's private LDS stripe
    f16* P = &ldsP[wave * 16 * PSTRIDE];
#pragma unroll
    for (int jt = 0; jt < 12; ++jt) {
#pragma unroll
      for (int r = 0; r < 8; ++r)
        P[(rbase + r) * PSTRIDE + jt * 16 + col] = (f16)(s[jt][r] * rsum[r]);
    }

    // O = P(16x192) x V(192x32): two 16-col tiles, K in chunks of 32.
    // B fragments contiguous from transposed V.
#pragma unroll
    for (int ct = 0; ct < 2; ++ct) {
      v8f o = {};
#pragma unroll
      for (int kt = 0; kt < 6; ++kt) {
        v16h ap = load_a_frag(P, PSTRIDE, 0, kt * 32, lane);
        v16h bv = load_b_frag(ldsVT, PSTRIDE, ct * 16, kt * 32, lane);
        o = __builtin_amdgcn_wmma_f32_16x16x32_f16(
            false, ap, false, bv, (short)0, o, false, false);
      }
      // gate + store f16
#pragma unroll
      for (int r = 0; r < 8; ++r) {
        int i = i0 + rbase + r;
        int c = ct * 16 + col;
        size_t tok = (size_t)(n * LW + i);
        float g = (float)gf[tok * OUT_C + h * DH + c];
        out[tok * OUT_C + h * DH + c] = (f16)(o[r] * g);
      }
    }
  }
}

// ---------------------------------------------------------------------------
// Host launcher
// ---------------------------------------------------------------------------
extern "C" void kernel_launch(void* const* d_in, const int* in_sizes, int n_in,
                              void* d_out, int out_size, void* d_ws, size_t ws_size,
                              hipStream_t stream) {
  const float* com2d = (const float*)d_in[0];
  const float* dist  = (const float*)d_in[1];
  const float* ln_w  = (const float*)d_in[2];
  const float* ln_b  = (const float*)d_in[3];
  const float* Wq = (const float*)d_in[4];  const float* bq = (const float*)d_in[5];
  const float* Wk = (const float*)d_in[6];  const float* bk = (const float*)d_in[7];
  const float* Wv = (const float*)d_in[8];  const float* bv = (const float*)d_in[9];
  const float* Wg = (const float*)d_in[10]; const float* bg = (const float*)d_in[11];
  const float* Wf = (const float*)d_in[12]; const float* bf = (const float*)d_in[13];
  float* out = (float*)d_out;

  char* ws = (char*)d_ws;
  const size_t act_bytes = (size_t)NTOK * OUT_C * sizeof(f16);   // 18.87 MB
  const size_t wmat = (size_t)OUT_C * IN_C;                      // 65536 elems
  f16*   xn   = (f16*)ws;                      ws += act_bytes;
  f16*   qf   = (f16*)ws;                      ws += act_bytes;
  f16*   kf   = (f16*)ws;                      ws += act_bytes;
  f16*   vf   = (f16*)ws;                      ws += act_bytes;
  f16*   gf   = (f16*)ws;                      ws += act_bytes;
  f16*   aout = (f16*)ws;                      ws += act_bytes;
  float* coefs = (float*)ws;                   ws += (size_t)NTOK * sizeof(float);
  f16*   wq16 = (f16*)ws;                      ws += wmat * sizeof(f16);
  f16*   wk16 = (f16*)ws;                      ws += wmat * sizeof(f16);
  f16*   wv16 = (f16*)ws;                      ws += wmat * sizeof(f16);
  f16*   wg16 = (f16*)ws;                      ws += wmat * sizeof(f16);
  f16*   wf16 = (f16*)ws;                      ws += wmat * sizeof(f16);
  (void)ws_size; (void)n_in; (void)in_sizes; (void)out_size;

  // Weight conversion f32 -> f16
  {
    int n = (int)wmat, blk = 256, grd = (n + blk - 1) / blk;
    cvt_f16_kernel<<<grd, blk, 0, stream>>>(Wq, wq16, n);
    cvt_f16_kernel<<<grd, blk, 0, stream>>>(Wk, wk16, n);
    cvt_f16_kernel<<<grd, blk, 0, stream>>>(Wv, wv16, n);
    cvt_f16_kernel<<<grd, blk, 0, stream>>>(Wg, wg16, n);
    cvt_f16_kernel<<<grd, blk, 0, stream>>>(Wf, wf16, n);
  }
  // Distance coefficient (softmax scale folded in)
  coef_kernel<<<(NTOK + 255) / 256, 256, 0, stream>>>(dist, coefs, NTOK);
  // LayerNorm
  ln_kernel<<<NTOK / 8, 256, 0, stream>>>(com2d, ln_w, ln_b, xn, NTOK);

  // Projections (f16 out): grid = (M/128, N/64) = (288, 4)
  dim3 ggrid(NTOK / 128, OUT_C / 64);
  gemm_wmma_kernel<<<ggrid, 256, 0, stream>>>(xn, wq16, bq, qf, nullptr, NTOK, OUT_C, IN_C);
  gemm_wmma_kernel<<<ggrid, 256, 0, stream>>>(xn, wk16, bk, kf, nullptr, NTOK, OUT_C, IN_C);
  gemm_wmma_kernel<<<ggrid, 256, 0, stream>>>(xn, wv16, bv, vf, nullptr, NTOK, OUT_C, IN_C);
  gemm_wmma_kernel<<<ggrid, 256, 0, stream>>>(xn, wg16, bg, gf, nullptr, NTOK, OUT_C, IN_C);

  // Attention + gating
  attn_kernel<<<dim3(LW, NH), ATT_WAVES * 32, 0, stream>>>(qf, kf, vf, gf, coefs, aout);

  // Final projection (f32 out to d_out)
  gemm_wmma_kernel<<<ggrid, 256, 0, stream>>>(aout, wf16, bf, nullptr, out, NTOK, IN_C, OUT_C);
}
